// HyperdimensionalMemory_51049981280862
// MI455X (gfx1250) — matrix operand order repaired and checked by reading, not compiled, not run on hardware
//
#include <hip/hip_runtime.h>
#include <math.h>

// ---------------------------------------------------------------------------
// HyperdimensionalMemory for MI455X (gfx1250, wave32, WMMA).
// B=1024, H=1024, HD=10000 (padded to 10016 so K % 32 == 0 for bf16 WMMA).
// All four big GEMMs run on v_wmma_f32_16x16x32_bf16 (f32 accumulate).
// Block tile 256x32, K-step 32, 8 waves, each wave owns a 32x32 output tile
// (4 accumulators; A frags reused across 2 B frags -> 4 wmma / 8 ds_load).
// ---------------------------------------------------------------------------

typedef __bf16 bf16_t;
typedef __bf16  v16bf __attribute__((ext_vector_type(16)));
typedef float   v8f   __attribute__((ext_vector_type(8)));
typedef unsigned int u32x4 __attribute__((ext_vector_type(4)));
typedef unsigned int u32x2 __attribute__((ext_vector_type(2)));

#define DIM_B   1024
#define DIM_H   1024
#define DIM_HD  10000
#define DIM_HDP 10016   // padded to multiple of 32

union FragBF {
    v16bf v;
    u32x4 q[2];
    bf16_t e[16];
};

// ---------------------------------------------------------------------------
// row_inv_norm: out[r] = 1 / max(||src[r,:]||_2, clampv)
// ---------------------------------------------------------------------------
template<typename T>
__global__ __launch_bounds__(256)
void row_inv_norm(const T* __restrict__ src, int ld, int ncols,
                  float clampv, float* __restrict__ out) {
    __shared__ float red[256];
    const int r = blockIdx.x;
    const int tid = threadIdx.x;
    float ss = 0.0f;
    for (int c = tid; c < ncols; c += 256) {
        float v = (float)src[(size_t)r * ld + c];
        ss += v * v;
    }
    red[tid] = ss;
    __syncthreads();
    for (int s = 128; s > 0; s >>= 1) {
        if (tid < s) red[tid] += red[tid + s];
        __syncthreads();
    }
    if (tid == 0) out[r] = 1.0f / fmaxf(sqrtf(red[0]), clampv);
}

// ---------------------------------------------------------------------------
// xs[b,h] = bf16( x[b,h] * invnorm[h] )   (fold base-vector normalization
// into the LHS of GEMM1 since the scale depends only on the contraction idx)
// ---------------------------------------------------------------------------
__global__ __launch_bounds__(256)
void scale_convert_x(const float* __restrict__ x, const float* __restrict__ invn,
                     bf16_t* __restrict__ xs, int total) {
    int i = blockIdx.x * 256 + threadIdx.x;
    if (i < total) xs[i] = (bf16_t)(x[i] * invn[i & (DIM_H - 1)]);
}

// ---------------------------------------------------------------------------
// dst[r, 0:ncolsP] = bf16(src[r, 0:ncols]) with zero pad for c >= ncols
// ---------------------------------------------------------------------------
__global__ __launch_bounds__(256)
void convert_pad(const float* __restrict__ src, bf16_t* __restrict__ dst,
                 int ncols, int ncolsP, int nrows) {
    long long i = (long long)blockIdx.x * 256 + threadIdx.x;
    long long total = (long long)nrows * ncolsP;
    if (i >= total) return;
    int r = (int)(i / ncolsP);
    int c = (int)(i % ncolsP);
    dst[i] = (c < ncols) ? (bf16_t)src[(size_t)r * ncols + c] : (bf16_t)0.0f;
}

// ---------------------------------------------------------------------------
// Row softmax: attn[r,:] = softmax(sim[r,:]), n == 1024, block per row.
// ---------------------------------------------------------------------------
__global__ __launch_bounds__(256)
void softmax_rows(const float* __restrict__ sim, bf16_t* __restrict__ attn, int n) {
    __shared__ float red[256];
    const int r = blockIdx.x;
    const int tid = threadIdx.x;
    const float* row = sim + (size_t)r * n;

    float vals[4];
    float vmax = -INFINITY;
    #pragma unroll
    for (int i = 0; i < 4; ++i) {
        vals[i] = row[tid + i * 256];
        vmax = fmaxf(vmax, vals[i]);
    }
    red[tid] = vmax;
    __syncthreads();
    for (int s = 128; s > 0; s >>= 1) {
        if (tid < s) red[tid] = fmaxf(red[tid], red[tid + s]);
        __syncthreads();
    }
    const float m = red[0];
    __syncthreads();

    float lsum = 0.0f;
    #pragma unroll
    for (int i = 0; i < 4; ++i) {
        vals[i] = __expf(vals[i] - m);
        lsum += vals[i];
    }
    red[tid] = lsum;
    __syncthreads();
    for (int s = 128; s > 0; s >>= 1) {
        if (tid < s) red[tid] += red[tid + s];
        __syncthreads();
    }
    const float rinv = 1.0f / red[0];
    #pragma unroll
    for (int i = 0; i < 4; ++i)
        attn[(size_t)r * n + tid + i * 256] = (bf16_t)(vals[i] * rinv);
}

// ---------------------------------------------------------------------------
// bf16 WMMA GEMM: C[M,N] = A[M,K] * B   (B row-major K x N if !BT,
//                                        B row-major N x K if BT)
// EPI 0: store bf16 C
// EPI 1: store f32  C * e0[row]*e0[col]*e1[col]      (cosine-sim * importance)
// EPI 2: store f32  C + e1[col] + e2[row*ldc+col]    (+b_dec, +x residual)
// ---------------------------------------------------------------------------
template<bool BT, int EPI>
__global__ __launch_bounds__(256)
void gemm_bf16_wmma(const bf16_t* __restrict__ A, int lda,
                    const bf16_t* __restrict__ Bm, int ldb,
                    void* __restrict__ Cout, int ldc, int K,
                    const float* __restrict__ e0,
                    const float* __restrict__ e1,
                    const float* __restrict__ e2) {
    constexpr int BM = 256, BN = 32, BK = 32;
    constexpr int LDSA = 40, LDSB = 40;           // bf16 elems; 80B rows (16B aligned)
    __shared__ bf16_t As[BM * LDSA];
    __shared__ bf16_t Bs[BN * LDSB];

    const int tid  = threadIdx.x;
    const int lane = tid & 31;
    const int wave = tid >> 5;
    const int wm   = wave * 32;                   // wave M offset in block tile
    const int m0   = blockIdx.y * BM;
    const int n0   = blockIdx.x * BN;

    // WMMA fragment lane geometry (contiguous k = 16h + s in A and B)
    const int lh   = (lane < 16) ? 0 : 1;
    const int l15  = lane & 15;
    const int kb0  = lh ? 16 : 0;                 // A frag: K values 16h..16h+7
    const int kb1  = lh ? 24 : 8;                 // A frag: K values 16h+8..16h+15

    v8f acc[2][2];
    #pragma unroll
    for (int f = 0; f < 2; ++f)
        #pragma unroll
        for (int j = 0; j < 2; ++j)
            acc[f][j] = (v8f){0.f,0.f,0.f,0.f,0.f,0.f,0.f,0.f};

    for (int k0 = 0; k0 < K; k0 += BK) {
        // ---- stage A tile (256x32 bf16): 64B per thread --------------------
        const bf16_t* gpA = A + (size_t)(m0 + tid) * lda + k0;
        u32x4 a0 = *(const u32x4*)(gpA);
        u32x4 a1 = *(const u32x4*)(gpA + 8);
        u32x4 a2 = *(const u32x4*)(gpA + 16);
        u32x4 a3 = *(const u32x4*)(gpA + 24);
        __builtin_prefetch(gpA + BK, 0, 1);       // global_prefetch_b8 (speculative)
        bf16_t* spA = &As[tid * LDSA];
        *(u32x4*)(spA)      = a0;
        *(u32x4*)(spA + 8)  = a1;
        *(u32x4*)(spA + 16) = a2;
        *(u32x4*)(spA + 24) = a3;

        // ---- stage B tile as Bs[n][k] (32x32 bf16) -------------------------
        if constexpr (!BT) {       // B is K x N row-major -> transpose into LDS
            const int bk = tid >> 3;
            const int bn = (tid & 7) * 4;
            union { u32x2 u; bf16_t e[4]; } bv;
            const bf16_t* gpB = Bm + (size_t)(k0 + bk) * ldb + n0 + bn;
            bv.u = *(const u32x2*)(gpB);
            __builtin_prefetch(gpB + (size_t)BK * ldb, 0, 1);
            #pragma unroll
            for (int j = 0; j < 4; ++j) Bs[(bn + j) * LDSB + bk] = bv.e[j];
        } else {                   // B is N x K row-major -> direct copy
            const int bn = tid >> 3;
            const int bk = (tid & 7) * 4;
            const bf16_t* gpB = Bm + (size_t)(n0 + bn) * ldb + k0 + bk;
            u32x2 u = *(const u32x2*)(gpB);
            __builtin_prefetch(gpB + BK, 0, 1);
            *(u32x2*)(&Bs[bn * LDSB + bk]) = u;
        }
        __syncthreads();

        // ---- load fragments from LDS --------------------------------------
        FragBF fa[2], fb[2];
        #pragma unroll
        for (int f = 0; f < 2; ++f) {
            const bf16_t* ap = &As[(wm + f * 16 + l15) * LDSA];
            fa[f].q[0] = *(const u32x4*)(ap + kb0);
            fa[f].q[1] = *(const u32x4*)(ap + kb1);
        }
        #pragma unroll
        for (int j = 0; j < 2; ++j) {
            const bf16_t* bp = &Bs[(j * 16 + l15) * LDSB + lh * 16];
            fb[j].q[0] = *(const u32x4*)(bp);
            fb[j].q[1] = *(const u32x4*)(bp + 8);
        }

        #pragma unroll
        for (int f = 0; f < 2; ++f)
            #pragma unroll
            for (int j = 0; j < 2; ++j)
                acc[f][j] = __builtin_amdgcn_wmma_f32_16x16x32_bf16(
                    false, fa[f].v, false, fb[j].v, (short)0, acc[f][j],
                    false, false);
        __syncthreads();
    }

    // ---- epilogue: C layout -> VGPR i holds row (i + (lane<16?0:8)) --------
    const int rb0 = m0 + wm + ((lane < 16) ? 0 : 8);

    #pragma unroll
    for (int f = 0; f < 2; ++f) {
        #pragma unroll
        for (int j = 0; j < 2; ++j) {
            const int col = n0 + j * 16 + l15;
            const int rbase = rb0 + f * 16;
            #pragma unroll
            for (int i = 0; i < 8; ++i) {
                const int row = rbase + i;
                const float v = acc[f][j][i];
                if constexpr (EPI == 0) {
                    ((bf16_t*)Cout)[(size_t)row * ldc + col] = (bf16_t)v;
                } else if constexpr (EPI == 1) {
                    ((float*)Cout)[(size_t)row * ldc + col] =
                        v * e0[row] * e0[col] * e1[col];
                } else {
                    ((float*)Cout)[(size_t)row * ldc + col] =
                        v + e1[col] + e2[(size_t)row * ldc + col];
                }
            }
        }
    }
}

// ---------------------------------------------------------------------------
// Host launcher
// ---------------------------------------------------------------------------
extern "C" void kernel_launch(void* const* d_in, const int* in_sizes, int n_in,
                              void* d_out, int out_size, void* d_ws, size_t ws_size,
                              hipStream_t stream) {
    const float* x          = (const float*)d_in[0];   // [1024,1,1024]
    const float* importance = (const float*)d_in[1];   // [1024]
    const float* base       = (const float*)d_in[2];   // [1024,10000]
    const float* W_dec      = (const float*)d_in[3];   // [1024,10000]
    const float* b_dec      = (const float*)d_in[4];   // [1024]
    float* out              = (float*)d_out;           // [1024,1,1024]

    // ---- workspace carve-up (all offsets 256B aligned) ---------------------
    char* w = (char*)d_ws;
    size_t off = 0;
    auto carve = [&](size_t bytes) -> char* {
        char* p = w + off;
        off += (bytes + 255) & ~(size_t)255;
        return p;
    };
    float*  invn  = (float*) carve(DIM_H * sizeof(float));            // 1/||base row||
    float*  qninv = (float*) carve(DIM_B * sizeof(float));            // 1/max(||enc||,eps)
    bf16_t* xs    = (bf16_t*)carve((size_t)DIM_B * DIM_H   * 2);      // scaled x, bf16
    bf16_t* baseB = (bf16_t*)carve((size_t)DIM_H * DIM_HDP * 2);      // base, bf16 padded
    bf16_t* wB    = (bf16_t*)carve((size_t)DIM_H * DIM_HDP * 2);      // W_dec, bf16 padded
    bf16_t* encB  = (bf16_t*)carve((size_t)DIM_B * DIM_HDP * 2);      // enc, bf16 padded
    float*  simF  = (float*) carve((size_t)DIM_B * DIM_B * 4);        // cosine sims
    bf16_t* attnB = (bf16_t*)carve((size_t)DIM_B * DIM_B * 2);        // softmax, bf16
    bf16_t* rhB   = (bf16_t*)carve((size_t)DIM_B * DIM_HDP * 2);      // retrieved_hyper
    (void)ws_size; (void)in_sizes; (void)n_in; (void)out_size;

    const dim3 blk(256);
    const dim3 gWide(DIM_HDP / 32, DIM_B / 256);   // (313, 4) N=10016 GEMMs
    const dim3 gSq  (DIM_B   / 32, DIM_B / 256);   // (32, 4)  N=1024  GEMMs
    const int  padBlocks = (DIM_H * DIM_HDP) / 256;

    // 1) invnorm[h] = 1/max(||base[h,:]||, 1e-12)
    row_inv_norm<float><<<DIM_H, blk, 0, stream>>>(base, DIM_HD, DIM_HD, 1e-12f, invn);
    // 2) xs = bf16(x * invnorm[h])  (folds base normalization into GEMM1 LHS)
    scale_convert_x<<<(DIM_B * DIM_H) / 256, blk, 0, stream>>>(x, invn, xs, DIM_B * DIM_H);
    // 3) bf16 + pad conversions of the two 40MB operands
    convert_pad<<<padBlocks, blk, 0, stream>>>(base,  baseB, DIM_HD, DIM_HDP, DIM_H);
    convert_pad<<<padBlocks, blk, 0, stream>>>(W_dec, wB,    DIM_HD, DIM_HDP, DIM_H);
    // 4) GEMM1 (NN): enc[B,HDP] = xs[B,H] @ baseB[H,HDP]
    gemm_bf16_wmma<false, 0><<<gWide, blk, 0, stream>>>(
        xs, DIM_H, baseB, DIM_HDP, encB, DIM_HDP, DIM_H, nullptr, nullptr, nullptr);
    // 5) qninv[b] = 1/max(||enc[b,:]||, 1e-8)   (pad cols are zero)
    row_inv_norm<bf16_t><<<DIM_B, blk, 0, stream>>>(encB, DIM_HDP, DIM_HDP, 1e-8f, qninv);
    // 6) GEMM2 (NT): sim = (enc @ enc^T) * qninv_r * qninv_c * importance_c
    gemm_bf16_wmma<true, 1><<<gSq, blk, 0, stream>>>(
        encB, DIM_HDP, encB, DIM_HDP, simF, DIM_B, DIM_HDP, qninv, importance, nullptr);
    // 7) attn = softmax(sim) rows
    softmax_rows<<<DIM_B, blk, 0, stream>>>(simF, attnB, DIM_B);
    // 8) GEMM3 (NN): rh[B,HDP] = attn[B,B] @ enc[B,HDP]
    gemm_bf16_wmma<false, 0><<<gWide, blk, 0, stream>>>(
        attnB, DIM_B, encB, DIM_HDP, rhB, DIM_HDP, DIM_B, nullptr, nullptr, nullptr);
    // 9) GEMM4 (NT): out = x + rh @ W_dec^T + b_dec
    gemm_bf16_wmma<true, 2><<<gSq, blk, 0, stream>>>(
        rhB, DIM_HDP, wB, DIM_HDP, out, DIM_H, DIM_HDP, nullptr, b_dec, x);
}
